// AugmentedModel_20942260535570
// MI455X (gfx1250) — matrix operand order repaired and checked by reading, not compiled, not run on hardware
//
#include <hip/hip_runtime.h>
#include <hip/hip_bf16.h>

// Problem dimensions (fixed by the reference)
#define T_STEPS 128
#define B_TOT   128
#define S_DIM   512
#define A_DIM   8

typedef __attribute__((ext_vector_type(16))) __bf16          v16bf;
typedef __attribute__((ext_vector_type(8)))  float           v8f;
typedef __attribute__((ext_vector_type(16))) unsigned short  v16u;
typedef __attribute__((ext_vector_type(8)))  unsigned short  v8u;

__device__ __forceinline__ unsigned short f2bf(float f) {
    unsigned u = __builtin_bit_cast(unsigned, f);
    unsigned r = u + 0x7fffu + ((u >> 16) & 1u);   // round-to-nearest-even
    return (unsigned short)(r >> 16);
}

// Pre-pass: Pb[a][s'][s] = bf16(exp(log_trans[a][s][s'])) — transposed, k contiguous.
__global__ void prep_exp_trans_bf16(const float* __restrict__ lt,
                                    unsigned short* __restrict__ pb, int n) {
    int i = blockIdx.x * blockDim.x + threadIdx.x;
    if (i < n) {
        int pa = i / (S_DIM * S_DIM);
        int pk = (i / S_DIM) & (S_DIM - 1);
        int pn = i & (S_DIM - 1);
        pb[((size_t)pa * S_DIM + pn) * S_DIM + pk] = f2bf(__expf(lt[i]));
    }
}

// One workgroup = 16 batch chains for all T+1 steps. 512 threads = 16 wave32s.
__global__ __launch_bounds__(512)
void hmm_forward_wmma(const int* __restrict__ regime,
                      const int* __restrict__ seq_o,
                      const int* __restrict__ seq_r,
                      const int* __restrict__ seq_d,
                      const int* __restrict__ seq_a,
                      const float* __restrict__ log_init,
                      const float* __restrict__ log_emit_o,   // [NO,S]
                      const float* __restrict__ log_emit_r,   // [NR,S]
                      const float* __restrict__ log_emit_d,   // [2,S]
                      const float* __restrict__ log_policy,   // [A,S]
                      const unsigned short* __restrict__ Pb,  // [A,S,S] bf16
                      float* __restrict__ out) {
    extern __shared__ char smem_raw[];
    float* bufA   = (float*)smem_raw;                 // 16*512 f32
    float* bufB   = bufA + 16 * S_DIM;                // 16*512 f32
    unsigned short* wbf = (unsigned short*)(bufB + 16 * S_DIM); // 16*512 bf16
    float* cinv   = (float*)(wbf + 16 * S_DIM);       // 16
    float* logacc = cinv + 16;                        // 16
    int*   meta   = (int*)(logacc + 16);              // 7*16 ints
    int *om = meta, *rm = meta + 16, *dm = meta + 32, *wasm = meta + 48;
    int *am = meta + 64, *regm = meta + 80, *dstate = meta + 96;

    const int tid  = threadIdx.x;
    const int bg   = blockIdx.x;      // batch group (16 chains)
    const int lane = tid & 31;
    const int wv   = tid >> 5;        // wave id 0..15

    if (tid < 16) {
        logacc[tid] = 0.0f;
        dstate[tid] = 0;
        regm[tid]   = regime[bg * 16 + tid];
    }
    // p0[s] = exp(log_init[s]) for every chain
    for (int i = tid; i < 16 * S_DIM; i += 512)
        bufA[i] = __expf(log_init[i & (S_DIM - 1)]);
    __syncthreads();

    float* pc = bufA;   // current alpha (scaled)
    float* pn = bufB;   // scratch for w, then next alpha

    for (int t = 0; t <= T_STEPS; ++t) {
        // ---- phase 0: per-chain metadata for this step ----
        if (tid < 16) {
            int b   = bg * 16 + tid;
            om[tid] = seq_o[t * B_TOT + b];
            rm[tid] = seq_r[t * B_TOT + b];
            int dc  = (t < T_STEPS) ? seq_d[t * B_TOT + b] : 1;  // last step forced done
            int was = dstate[tid];
            int dnw = (was > dc) ? was : dc;                     // cummax
            dstate[tid] = dnw;
            dm[tid]   = dnw;
            wasm[tid] = was;
            am[tid]   = (t < T_STEPS) ? seq_a[t * B_TOT + b] : 0;
        }
        __syncthreads();

        // ---- phase 1: w[m][s] = alpha[m][s] * exp(emission) ----
        for (int i = tid; i < 16 * S_DIM; i += 512) {
            int m = i >> 9;
            int s = i & (S_DIM - 1);
            float e = log_emit_o[om[m] * S_DIM + s]
                    + log_emit_r[rm[m] * S_DIM + s]
                    + log_emit_d[dm[m] * S_DIM + s];
            if (wasm[m]) e = 0.0f;                       // already done before this step
            float pa = log_policy[am[m] * S_DIM + s];
            if (dm[m] | regm[m]) pa = 0.0f;              // done OR regime==1 masks policy
            pn[i] = pc[i] * __expf(e + pa);
        }
        __syncthreads();

        // ---- phase 2: c[m] = sum_s w[m][s]; wave wv owns row wv ----
        {
            float acc = 0.0f;
            const int m = wv;
            #pragma unroll
            for (int j = 0; j < 16; ++j) acc += pn[m * S_DIM + lane + 32 * j];
            #pragma unroll
            for (int off = 16; off >= 1; off >>= 1) acc += __shfl_xor(acc, off, 32);
            if (lane == 0) {
                cinv[m]    = 1.0f / acc;
                logacc[m] += __logf(acc);                // log_prob accumulation
            }
        }
        __syncthreads();

        if (t == T_STEPS) break;   // final transition is unused by the reference

        // ---- phase 3: normalized bf16 A-matrix staging ----
        for (int i = tid; i < 16 * S_DIM; i += 512)
            wbf[i] = f2bf(pn[i] * cinv[i >> 9]);
        __syncthreads();

        // ---- phase 4: alpha_next[m][n] = sum_k A[m][k] * P[a_m][k][n] via WMMA.
        //      All 8 actions always processed (row-masked); (kc, a) flattened and
        //      software-pipelined: B fragments for iteration i+1 are issued before
        //      the WMMAs of iteration i consume their (already-arrived) fragments.
        {
            const int myA  = am[lane & 15];              // this lane's A-row action
            const int n0   = wv * 32;                    // two 16-wide N tiles per wave
            const int arow = lane & 15;
            const int koff = (lane & 16) ? 8 : 0;        // ISA 16-bit A layout
            const v16u zu = {};
            v8f acc0 = {}, acc1 = {};

            // per-lane B base: Pb[(a*S + n0 + (lane&15))*S + k0 + koff] = bln + a*S*S + k0
            const unsigned short* bln = Pb + (size_t)(n0 + (lane & 15)) * S_DIM + koff;

            // stage 0 preload: A fragment (kc=0), B fragments (a=0, kc=0)
            v8u alo = *(const v8u*)&wbf[arow * S_DIM + koff];
            v8u ahi = *(const v8u*)&wbf[arow * S_DIM + 16 + koff];
            v16u au = __builtin_shufflevector(alo, ahi,
                      0,1,2,3,4,5,6,7,8,9,10,11,12,13,14,15);
            v16u b0 = __builtin_shufflevector(*(const v8u*)bln, *(const v8u*)(bln + 16),
                      0,1,2,3,4,5,6,7,8,9,10,11,12,13,14,15);
            v16u b1 = __builtin_shufflevector(*(const v8u*)(bln + 16 * S_DIM),
                      *(const v8u*)(bln + 16 * S_DIM + 16),
                      0,1,2,3,4,5,6,7,8,9,10,11,12,13,14,15);

            #pragma unroll 8
            for (int it = 0; it < 16 * A_DIM; ++it) {
                const int a   = it & (A_DIM - 1);
                const int na  = (it + 1) & (A_DIM - 1);
                const int nk0 = ((it + 1) >> 3) * 32;
                // prefetch next B fragments (tail overrun stays inside Pb, unused)
                const unsigned short* nbp =
                    bln + (size_t)na * S_DIM * S_DIM + nk0;
                v16u nb0 = __builtin_shufflevector(*(const v8u*)nbp, *(const v8u*)(nbp + 16),
                           0,1,2,3,4,5,6,7,8,9,10,11,12,13,14,15);
                v16u nb1 = __builtin_shufflevector(*(const v8u*)(nbp + 16 * S_DIM),
                           *(const v8u*)(nbp + 16 * S_DIM + 16),
                           0,1,2,3,4,5,6,7,8,9,10,11,12,13,14,15);
                // prefetch next A fragment at the action-loop boundary
                v16u nau = au;
                if (na == 0) {
                    v8u nlo = *(const v8u*)&wbf[arow * S_DIM + nk0 + koff];
                    v8u nhi = *(const v8u*)&wbf[arow * S_DIM + nk0 + 16 + koff];
                    nau = __builtin_shufflevector(nlo, nhi,
                          0,1,2,3,4,5,6,7,8,9,10,11,12,13,14,15);
                }
                v16u  amsk = (myA == a) ? au : zu;       // zero rows of other actions
                v16bf af   = __builtin_bit_cast(v16bf, amsk);
                acc0 = __builtin_amdgcn_wmma_f32_16x16x32_bf16(
                           false, af, false, __builtin_bit_cast(v16bf, b0),
                           (short)0, acc0, false, false);
                acc1 = __builtin_amdgcn_wmma_f32_16x16x32_bf16(
                           false, af, false, __builtin_bit_cast(v16bf, b1),
                           (short)0, acc1, true, false); // reuse_a: same A as prev
                au = nau; b0 = nb0; b1 = nb1;
            }
            // D layout: VGPR r -> m = r + (lane&16 ? 8 : 0), n = n0 + (lane&15)
            const int mb = (lane & 16) ? 8 : 0;
            const int nc = lane & 15;
            #pragma unroll
            for (int r = 0; r < 8; ++r) {
                pn[(mb + r) * S_DIM + n0 + nc]      = acc0[r];
                pn[(mb + r) * S_DIM + n0 + 16 + nc] = acc1[r];
            }
        }
        __syncthreads();
        float* tswap = pc; pc = pn; pn = tswap;
    }

    if (tid < 16) out[bg * 16 + tid] = logacc[tid];
}

extern "C" void kernel_launch(void* const* d_in, const int* in_sizes, int n_in,
                              void* d_out, int out_size, void* d_ws, size_t ws_size,
                              hipStream_t stream) {
    const int*   regime     = (const int*)  d_in[0];
    const int*   seq_o      = (const int*)  d_in[1];
    const int*   seq_r      = (const int*)  d_in[2];
    const int*   seq_d      = (const int*)  d_in[3];
    const int*   seq_a      = (const int*)  d_in[4];
    const float* log_init   = (const float*)d_in[5];
    const float* log_trans  = (const float*)d_in[6];
    const float* log_emit_o = (const float*)d_in[7];
    const float* log_emit_r = (const float*)d_in[8];
    const float* log_emit_d = (const float*)d_in[9];
    const float* log_policy = (const float*)d_in[10];
    float* out = (float*)d_out;

    // Workspace: bf16 exp(log_trans), 8*512*512*2 = 4 MB (L2-resident)
    unsigned short* Pb = (unsigned short*)d_ws;
    const int nP = A_DIM * S_DIM * S_DIM;
    prep_exp_trans_bf16<<<(nP + 255) / 256, 256, 0, stream>>>(log_trans, Pb, nP);

    const size_t shmem = (size_t)(16 * S_DIM * 4) * 2   // alpha double buffer
                       + (size_t)(16 * S_DIM * 2)       // bf16 A staging
                       + 2 * 16 * sizeof(float)         // cinv, logacc
                       + 7 * 16 * sizeof(int);          // metadata
    hmm_forward_wmma<<<B_TOT / 16, 512, shmem, stream>>>(
        regime, seq_o, seq_r, seq_d, seq_a, log_init,
        log_emit_o, log_emit_r, log_emit_d, log_policy, Pb, out);
}
